// AtomicLinear_30064771072600
// MI455X (gfx1250) — matrix operand order repaired and checked by reading, not compiled, not run on hardware
//
#include <hip/hip_runtime.h>
#include <stdint.h>

// ---------------------------------------------------------------------------
// out[4096][2048] = X[4096][2048] . W[2048][2048]^T + bias
// (index permutation in the reference cancels: same sigma on both K dims)
// Pass 1: f32 -> bf16 into d_ws.  Pass 2: double-buffered LDS WMMA GEMM with
// GLOBAL_LOAD_ASYNC_TO_LDS_B128 (no VGPR staging -> no spills).
// ---------------------------------------------------------------------------

#define BATCH 4096
#define NOUT  2048
#define KDIM  2048

#define BM 128          // block tile M
#define BN 128          // block tile N
#define BK 64           // K tile staged in LDS
#define LDT 72          // padded LDS row stride (bf16 elems); 144B, 16B-aligned

#define BUF_SHORTS ((BM + BN) * LDT)          // 18432 shorts per buffer
#define BUF_BYTES  (BUF_SHORTS * 2)           // 36864 B; x2 buffers = 72 KB

typedef __attribute__((ext_vector_type(16))) __bf16          bf16x16;
typedef __attribute__((ext_vector_type(8)))  float           f32x8;
typedef __attribute__((ext_vector_type(8)))  unsigned short  u16x8;
typedef __attribute__((ext_vector_type(4)))  int             v4i_t;

struct Frag32B { u16x8 lo, hi; };   // 32 bytes == bf16x16

static __device__ __forceinline__ unsigned short f2bf_rne(float f) {
    unsigned u = __builtin_bit_cast(unsigned, f);
    u += 0x7FFFu + ((u >> 16) & 1u);            // round-to-nearest-even
    return (unsigned short)(u >> 16);
}

// Async global->LDS 16B copy (ASYNCcnt-tracked, no VGPR staging).
// Builtin signature (from hipcc diagnostic): (int4 AS1*, int4 AS3*, Iimm, Iimm)
static __device__ __forceinline__ void async_copy_b128(unsigned short* lds_ptr,
                                                       const unsigned short* gptr) {
#if __has_builtin(__builtin_amdgcn_global_load_async_to_lds_b128)
    __builtin_amdgcn_global_load_async_to_lds_b128(
        (__attribute__((address_space(1))) v4i_t*)(uintptr_t)gptr,
        (__attribute__((address_space(3))) v4i_t*)(unsigned)(uintptr_t)lds_ptr,
        /*offset=*/0, /*cpol=*/0);
#else
    asm volatile("global_load_async_to_lds_b128 %0, %1, off"
                 :: "v"((unsigned)(uintptr_t)lds_ptr), "v"(gptr)
                 : "memory");
#endif
}

static __device__ __forceinline__ void wait_async_le8() {
#if __has_builtin(__builtin_amdgcn_s_wait_asynccnt)
    __builtin_amdgcn_s_wait_asynccnt(8);
#else
    asm volatile("s_wait_asynccnt 0x8" ::: "memory");
#endif
}
static __device__ __forceinline__ void wait_async_le0() {
#if __has_builtin(__builtin_amdgcn_s_wait_asynccnt)
    __builtin_amdgcn_s_wait_asynccnt(0);
#else
    asm volatile("s_wait_asynccnt 0x0" ::: "memory");
#endif
}

// ---------------------------------------------------------------------------
// Pass 1: f32 -> bf16 for both X and W (4 elems / thread, 16B load, 8B store)
// ---------------------------------------------------------------------------
__global__ __launch_bounds__(256) void cvt_bf16(
    const float* __restrict__ x, const float* __restrict__ w,
    unsigned short* __restrict__ xb, unsigned short* __restrict__ wb) {
    const long nx = (long)BATCH * KDIM;
    long i = ((long)blockIdx.x * 256 + threadIdx.x) * 4;
    const float* src;
    unsigned short* dst;
    long j;
    if (i < nx) { src = x; dst = xb; j = i; }
    else        { src = w; dst = wb; j = i - nx; }
    float4 v = *(const float4*)(src + j);
    unsigned long long p =
        (unsigned long long)f2bf_rne(v.x)
      | ((unsigned long long)f2bf_rne(v.y) << 16)
      | ((unsigned long long)f2bf_rne(v.z) << 32)
      | ((unsigned long long)f2bf_rne(v.w) << 48);
    *(unsigned long long*)(dst + j) = p;
}

// ---------------------------------------------------------------------------
// Pass 2: tiled GEMM, 256 threads = 8 wave32s, 128x128 tile, BK=64, LDS
// double-buffered via async global->LDS copies. Wave grid 4(M) x 2(N): each
// wave computes 32x64 = 2x4 WMMA 16x16 accumulators.
// ---------------------------------------------------------------------------
__global__ __launch_bounds__(256) void gemm_bf16_wmma(
    const unsigned short* __restrict__ Xb,   // [BATCH][KDIM] bf16
    const unsigned short* __restrict__ Wb,   // [NOUT][KDIM]  bf16 (K contiguous)
    const float* __restrict__ bias,          // [NOUT]
    float* __restrict__ out) {               // [BATCH][NOUT]

    // Two buffers; within a buffer: A tile then B tile, both row-pitch LDT.
    __shared__ unsigned short Sm[2 * BUF_SHORTS];

    const int tid  = threadIdx.x;
    const int bm   = blockIdx.x * BM;        // batch origin
    const int bn   = blockIdx.y * BN;        // out-feature origin

    // global->LDS cooperative mapping: 16B per thread, 8 chunks per 64-elem row
    const int lrow   = tid >> 3;             // 0..31 (4 row passes of 32)
    const int lchunk = tid & 7;              // 0..7

    // wave / lane decomposition
    const int wid  = tid >> 5;               // 0..7
    const int lane = tid & 31;
    const int half = lane >> 4;              // 0/1
    const int l16  = lane & 15;
    const int wm   = (wid & 3) * 32;         // wave M origin within tile
    const int wn   = (wid >> 2) * 64;        // wave N origin within tile

    // per-thread destination offsets within a buffer (element units)
    unsigned a_off[4], b_off[4];
    #pragma unroll
    for (int p = 0; p < 4; ++p) {
        a_off[p] = (lrow + 32 * p) * LDT + lchunk * 8;
        b_off[p] = a_off[p] + BM * LDT;
    }

    f32x8 acc[2][4];
    #pragma unroll
    for (int mt = 0; mt < 2; ++mt)
        #pragma unroll
        for (int nt = 0; nt < 4; ++nt)
            acc[mt][nt] = (f32x8){0.f, 0.f, 0.f, 0.f, 0.f, 0.f, 0.f, 0.f};

    // issue K-tile 0 into buffer 0
    {
        const int ko = lchunk * 8;
        #pragma unroll
        for (int p = 0; p < 4; ++p)
            async_copy_b128(&Sm[a_off[p]],
                            Xb + (long)(bm + lrow + 32 * p) * KDIM + ko);
        #pragma unroll
        for (int p = 0; p < 4; ++p)
            async_copy_b128(&Sm[b_off[p]],
                            Wb + (long)(bn + lrow + 32 * p) * KDIM + ko);
    }

    const int NK = KDIM / BK;                // 32 K-tiles
    for (int kt = 0; kt < NK; ++kt) {
        const int cur = kt & 1;

        // issue next tile into the other buffer (it was last read in kt-1;
        // the trailing barrier of kt-1 makes this safe)
        if (kt + 1 < NK) {
            const unsigned base = (cur ^ 1) * BUF_SHORTS;
            const int ko = (kt + 1) * BK + lchunk * 8;
            #pragma unroll
            for (int p = 0; p < 4; ++p)
                async_copy_b128(&Sm[base + a_off[p]],
                                Xb + (long)(bm + lrow + 32 * p) * KDIM + ko);
            #pragma unroll
            for (int p = 0; p < 4; ++p)
                async_copy_b128(&Sm[base + b_off[p]],
                                Wb + (long)(bn + lrow + 32 * p) * KDIM + ko);
            wait_async_le8();   // >8 outstanding -> first 8 (tile kt) have landed
        } else {
            wait_async_le0();   // final tile: drain everything
        }
        __syncthreads();        // all waves' copies visible

        const unsigned short* As = &Sm[cur * BUF_SHORTS];
        const unsigned short* Bs = As + BM * LDT;

        // two 16x16x32 substeps per BK=64 tile
        #pragma unroll
        for (int ks = 0; ks < BK; ks += 32) {
            bf16x16 afrag[2], bfrag[4];

            // A fragment (16x32 bf16): lane = row m; K = half*8..+7 and +16..+23
            #pragma unroll
            for (int mt = 0; mt < 2; ++mt) {
                const unsigned short* p0 =
                    &As[(wm + mt * 16 + l16) * LDT + ks + half * 8];
                Frag32B f;
                f.lo = *(const u16x8*)(p0);
                f.hi = *(const u16x8*)(p0 + 16);
                afrag[mt] = __builtin_bit_cast(bf16x16, f);
            }
            // B fragment (32x16 bf16): lane = col n; K = half*8..+7 and +16..+23
            #pragma unroll
            for (int nt = 0; nt < 4; ++nt) {
                const unsigned short* p0 =
                    &Bs[(wn + nt * 16 + l16) * LDT + ks + half * 8];
                Frag32B f;
                f.lo = *(const u16x8*)(p0);
                f.hi = *(const u16x8*)(p0 + 16);
                bfrag[nt] = __builtin_bit_cast(bf16x16, f);
            }

            #pragma unroll
            for (int mt = 0; mt < 2; ++mt)
                #pragma unroll
                for (int nt = 0; nt < 4; ++nt)
                    acc[mt][nt] = __builtin_amdgcn_wmma_f32_16x16x32_bf16(
                        /*neg_a=*/false, afrag[mt],
                        /*neg_b=*/false, bfrag[nt],
                        /*c_mod=*/(short)0, acc[mt][nt],
                        /*reuse_a=*/false, /*reuse_b=*/false);
        }
        __syncthreads();        // everyone done reading cur before kt+1 overwrites it
    }

    // epilogue: C/D layout -> lane l16 = N, VGPR v = M row (half*8 + v)
    #pragma unroll
    for (int nt = 0; nt < 4; ++nt) {
        const int gn = bn + wn + nt * 16 + l16;
        const float bv = bias[gn];
        #pragma unroll
        for (int mt = 0; mt < 2; ++mt) {
            const int gm = bm + wm + mt * 16 + half * 8;
            float* o = out + (long)gm * NOUT + gn;
            #pragma unroll
            for (int v = 0; v < 8; ++v)
                o[(long)v * NOUT] = acc[mt][nt][v] + bv;
        }
    }
}

// ---------------------------------------------------------------------------
extern "C" void kernel_launch(void* const* d_in, const int* in_sizes, int n_in,
                              void* d_out, int out_size, void* d_ws, size_t ws_size,
                              hipStream_t stream) {
    (void)in_sizes; (void)n_in; (void)out_size; (void)ws_size;
    const float* x    = (const float*)d_in[0];
    const float* w    = (const float*)d_in[1];
    const float* bias = (const float*)d_in[2];
    // d_in[3] (int64 permutation) intentionally unused: the same permutation is
    // applied to the K dim of both operands, so it cancels in the dot product.

    unsigned short* xb = (unsigned short*)d_ws;                    // 16 MB bf16 X
    unsigned short* wb = xb + (size_t)BATCH * KDIM;                //  8 MB bf16 W
    float* out = (float*)d_out;

    const long total4 = ((long)BATCH * KDIM + (long)NOUT * KDIM) / 4;
    cvt_bf16<<<(int)(total4 / 256), 256, 0, stream>>>(x, w, xb, wb);

    dim3 grid(BATCH / BM, NOUT / BN);   // 32 x 16 = 512 workgroups
    gemm_bf16_wmma<<<grid, 256, 0, stream>>>(xb, wb, bias, out);
}
